// BiLSTM_CRF_48086453846209
// MI455X (gfx1250) — matrix operand order repaired and checked by reading, not compiled, not run on hardware
//
#include <hip/hip_runtime.h>

typedef _Float16 h16_t;
typedef __attribute__((ext_vector_type(16))) _Float16 v16h;
typedef __attribute__((ext_vector_type(8)))  _Float16 v8h;
typedef __attribute__((ext_vector_type(8)))  float    v8f;

#define T_LEN   2048
#define E_DIM   256
#define H_DIM   256
#define G4H     1024   // 4*H
#define HID     512
#define K_TAGS  32
#define START_TAG 30
#define STOP_TAG  31
#define NEG_INF (-10000.0f)

// ---------------- prep kernels ----------------

__global__ void k_gather(const int* __restrict__ sent, const float* __restrict__ embed,
                         h16_t* __restrict__ X16) {
  const int t = blockIdx.x, e = threadIdx.x;
  X16[(size_t)t * E_DIM + e] = (h16_t)embed[(size_t)sent[t] * E_DIM + e];
}

__global__ void k_cvt(const float* __restrict__ src, h16_t* __restrict__ dst, int n) {
  int i = blockIdx.x * blockDim.x + threadIdx.x;
  if (i < n) dst[i] = (h16_t)src[i];
}

__global__ void k_bias(const float* __restrict__ bi, const float* __restrict__ bh,
                       float* __restrict__ b) {
  int i = blockIdx.x * blockDim.x + threadIdx.x;
  b[i] = bi[i] + bh[i];
}

// ---------------- WMMA GEMM: Z = X(2048x256) @ Wi^T(256x1024) + bias ----------------
// One wave per 16x16 C tile. A fragment: lane reads row M, two contiguous 8-half runs.
// B column N == Wi row N (contiguous) -> one 32B load per lane.

__global__ void k_gemm_z(const h16_t* __restrict__ X, const h16_t* __restrict__ Wi,
                         const float* __restrict__ bias, float* __restrict__ Z) {
  const int lane  = threadIdx.x & 31;
  const int wave  = threadIdx.x >> 5;
  const int ntile = blockIdx.x * 4 + wave;     // 0..63
  const int mtile = blockIdx.y;                // 0..127
  const int mrow  = mtile * 16 + (lane & 15);
  const int ncol  = ntile * 16 + (lane & 15);
  const int aoff  = (lane < 16) ? 0 : 16;
  const int boff  = (lane < 16) ? 0 : 16;
  v8f acc = {};
#pragma unroll
  for (int kc = 0; kc < 8; ++kc) {
    const int kb = kc * 32;
    v16h a;
    *((v8h*)&a)       = *(const v8h*)(X + (size_t)mrow * E_DIM + kb + aoff);
    *(((v8h*)&a) + 1) = *(const v8h*)(X + (size_t)mrow * E_DIM + kb + aoff + 8);
    v16h b = *(const v16h*)(Wi + (size_t)ncol * E_DIM + kb + boff);
    acc = __builtin_amdgcn_wmma_f32_16x16x32_f16(false, a, false, b, (short)0, acc, false, false);
  }
  const float bv    = bias[ncol];
  const int   mbase = mtile * 16 + ((lane < 16) ? 0 : 8);
#pragma unroll
  for (int v = 0; v < 8; ++v)
    Z[(size_t)(mbase + v) * G4H + ncol] = acc[v] + bv;
}

// ---------------- recurrent LSTM scan (the serial critical path) ----------------
// grid = 2 (dir), block = 1024 = 32 waves. Wh (f16) stays resident in VGPRs:
// wave w owns rows [32w, 32w+32): 2 row-tiles x 8 k-chunks = 128 A VGPRs.
// Per step: B = h broadcast from LDS, 16 WMMA, z -> LDS, 256 gate threads, repeat.

__global__ __launch_bounds__(1024)
void k_lstm(const h16_t* __restrict__ WhF, const h16_t* __restrict__ WhB,
            const float* __restrict__ Zf,  const float* __restrict__ Zb,
            const float* __restrict__ h0,  const float* __restrict__ c0,
            h16_t* __restrict__ H16) {
  const int dir  = blockIdx.x;
  const h16_t* Wh = dir ? WhB : WhF;
  const float* Zp = dir ? Zb  : Zf;
  const int tid  = threadIdx.x;
  const int lane = tid & 31, wave = tid >> 5;

  __shared__ __align__(64) h16_t sh_h[H_DIM];
  __shared__ __align__(64) float sh_z[G4H];

  // resident A fragments for rows [wave*32, wave*32+32)
  const int r0   = wave * 32 + (lane & 15);
  const int r1   = r0 + 16;
  const int aoff = (lane < 16) ? 0 : 16;
  v16h A0[8], A1[8];
#pragma unroll
  for (int kc = 0; kc < 8; ++kc) {
    const int kb = kc * 32 + aoff;
    *((v8h*)&A0[kc])       = *(const v8h*)(Wh + (size_t)r0 * H_DIM + kb);
    *(((v8h*)&A0[kc]) + 1) = *(const v8h*)(Wh + (size_t)r0 * H_DIM + kb + 8);
    *((v8h*)&A1[kc])       = *(const v8h*)(Wh + (size_t)r1 * H_DIM + kb);
    *(((v8h*)&A1[kc]) + 1) = *(const v8h*)(Wh + (size_t)r1 * H_DIM + kb + 8);
  }

  float c = 0.f, h = 0.f;
  if (tid < H_DIM) {
    h = h0[dir * H_DIM + tid];
    c = c0[dir * H_DIM + tid];
    sh_h[tid] = (h16_t)h;
  }
  __syncthreads();

  const int boff = (lane < 16) ? 0 : 16;
  for (int s = 0; s < T_LEN; ++s) {
    const int t = dir ? (T_LEN - 1 - s) : s;
    v8f d0 = {}, d1 = {};
#pragma unroll
    for (int kc = 0; kc < 8; ++kc) {
      v16h b = *(const v16h*)(&sh_h[kc * 32 + boff]);   // broadcast: identical per half-wave
      d0 = __builtin_amdgcn_wmma_f32_16x16x32_f16(false, A0[kc], false, b, (short)0, d0, false, false);
      d1 = __builtin_amdgcn_wmma_f32_16x16x32_f16(false, A1[kc], false, b, (short)0, d1, false, false);
    }
    // columns of D are identical; lanes 0 and 16 of each wave own distinct M rows
    if ((lane & 15) == 0) {
      const int mb = wave * 32 + ((lane < 16) ? 0 : 8);
#pragma unroll
      for (int v = 0; v < 8; ++v) {
        sh_z[mb + v]      = d0[v];
        sh_z[mb + 16 + v] = d1[v];
      }
    }
    __syncthreads();
    if (tid < H_DIM) {
      const size_t zb = (size_t)t * G4H;
      const float zi = sh_z[tid      ] + Zp[zb + tid      ];
      const float zf = sh_z[tid + 256] + Zp[zb + tid + 256];
      const float zg = sh_z[tid + 512] + Zp[zb + tid + 512];
      const float zo = sh_z[tid + 768] + Zp[zb + tid + 768];
      const float ig = 1.f / (1.f + __expf(-zi));
      const float fg = 1.f / (1.f + __expf(-zf));
      const float og = 1.f / (1.f + __expf(-zo));
      const float gg = tanhf(zg);
      c = fg * c + ig * gg;
      h = og * tanhf(c);
      sh_h[tid] = (h16_t)h;
      H16[(size_t)t * HID + dir * H_DIM + tid] = (h16_t)h;
      // prefetch next step's Z row (global_prefetch_b8)
      const int sn = (s + 1 < T_LEN) ? s + 1 : s;
      const int tn = dir ? (T_LEN - 1 - sn) : sn;
      __builtin_prefetch((const void*)(Zp + (size_t)tn * G4H + tid), 0, 0);
      __builtin_prefetch((const void*)(Zp + (size_t)tn * G4H + tid + 512), 0, 0);
    }
    __syncthreads();
  }
}

// ---------------- feats = [Hf|Hb](2048x512) @ W_out^T(512x32) + b_out ----------------

__global__ void k_feats(const h16_t* __restrict__ H16, const h16_t* __restrict__ Wo16,
                        const float* __restrict__ bo, float* __restrict__ feats) {
  const int lane  = threadIdx.x & 31;
  const int wave  = threadIdx.x >> 5;   // 2 waves -> 2 N tiles
  const int mtile = blockIdx.x;
  const int mrow  = mtile * 16 + (lane & 15);
  const int ncol  = wave * 16 + (lane & 15);
  const int aoff  = (lane < 16) ? 0 : 16;
  const int boff  = (lane < 16) ? 0 : 16;
  v8f acc = {};
#pragma unroll
  for (int kc = 0; kc < 16; ++kc) {
    const int kb = kc * 32;
    v16h a;
    *((v8h*)&a)       = *(const v8h*)(H16 + (size_t)mrow * HID + kb + aoff);
    *(((v8h*)&a) + 1) = *(const v8h*)(H16 + (size_t)mrow * HID + kb + aoff + 8);
    v16h b = *(const v16h*)(Wo16 + (size_t)ncol * HID + kb + boff);
    acc = __builtin_amdgcn_wmma_f32_16x16x32_f16(false, a, false, b, (short)0, acc, false, false);
  }
  const float bv    = bo[ncol];
  const int   mbase = mtile * 16 + ((lane < 16) ? 0 : 8);
#pragma unroll
  for (int v = 0; v < 8; ++v)
    feats[(size_t)(mbase + v) * K_TAGS + ncol] = acc[v] + bv;
}

// ---------------- Viterbi: single wave32, shuffle-based, 8-bit LDS backtracks ----------------

__global__ void k_viterbi(const float* __restrict__ feats, const float* __restrict__ trans,
                          float* __restrict__ out) {
  __shared__ unsigned char sbt[T_LEN * K_TAGS];   // 64 KB backtrack table
  const int i = threadIdx.x;                      // tag 0..31, one wave
  float tr[K_TAGS];
#pragma unroll
  for (int j = 0; j < K_TAGS; ++j) tr[j] = trans[i * K_TAGS + j];

  float fv = (i == START_TAG) ? 0.f : NEG_INF;
  for (int t = 0; t < T_LEN; ++t) {
    float best = -3.4e38f; int bj = 0;
#pragma unroll
    for (int j = 0; j < K_TAGS; ++j) {
      const float s = __shfl(fv, j, 32) + tr[j];
      if (s > best) { best = s; bj = j; }         // strict '>' => first max index
    }
    fv = best + feats[(size_t)t * K_TAGS + i];
    sbt[t * K_TAGS + i] = (unsigned char)bj;
  }
  const float term = fv + trans[STOP_TAG * K_TAGS + i];
  float best = -3.4e38f; int bi = 0;
#pragma unroll
  for (int j = 0; j < K_TAGS; ++j) {
    const float tv = __shfl(term, j, 32);
    if (tv > best) { best = tv; bi = j; }
  }
  if (i == 0) {
    out[0] = best;
    int tag = bi;
    for (int t = T_LEN - 1; t >= 0; --t) {
      out[1 + t] = (float)tag;
      tag = (int)sbt[t * K_TAGS + tag];
    }
  }
}

// ---------------- launcher ----------------

extern "C" void kernel_launch(void* const* d_in, const int* in_sizes, int n_in,
                              void* d_out, int out_size, void* d_ws, size_t ws_size,
                              hipStream_t stream) {
  const int*   sentence = (const int*)  d_in[0];
  const float* embed    = (const float*)d_in[1];
  const float* Wi_f     = (const float*)d_in[2];
  const float* Wh_f     = (const float*)d_in[3];
  const float* bi_f     = (const float*)d_in[4];
  const float* bh_f     = (const float*)d_in[5];
  const float* Wi_b     = (const float*)d_in[6];
  const float* Wh_b     = (const float*)d_in[7];
  const float* bi_b     = (const float*)d_in[8];
  const float* bh_b     = (const float*)d_in[9];
  const float* W_out    = (const float*)d_in[10];
  const float* b_out    = (const float*)d_in[11];
  const float* trans    = (const float*)d_in[12];
  const float* h0       = (const float*)d_in[13];
  const float* c0       = (const float*)d_in[14];

  char* p = (char*)d_ws;
  auto take = [&](size_t bytes) {
    char* r = p;
    p += (bytes + 255) & ~(size_t)255;
    return r;
  };
  h16_t* X16   = (h16_t*)take((size_t)T_LEN * E_DIM * 2);
  h16_t* WiF16 = (h16_t*)take((size_t)G4H * E_DIM * 2);
  h16_t* WiB16 = (h16_t*)take((size_t)G4H * E_DIM * 2);
  h16_t* WhF16 = (h16_t*)take((size_t)G4H * H_DIM * 2);
  h16_t* WhB16 = (h16_t*)take((size_t)G4H * H_DIM * 2);
  h16_t* Wo16  = (h16_t*)take((size_t)K_TAGS * HID * 2);
  float* biasF = (float*)take((size_t)G4H * 4);
  float* biasB = (float*)take((size_t)G4H * 4);
  float* Zf    = (float*)take((size_t)T_LEN * G4H * 4);
  float* Zb    = (float*)take((size_t)T_LEN * G4H * 4);
  h16_t* H16   = (h16_t*)take((size_t)T_LEN * HID * 2);
  float* feats = (float*)take((size_t)T_LEN * K_TAGS * 4);

  // precision prep
  k_gather<<<T_LEN, E_DIM, 0, stream>>>(sentence, embed, X16);
  const int nwi = G4H * E_DIM;
  k_cvt<<<(nwi + 255) / 256, 256, 0, stream>>>(Wi_f, WiF16, nwi);
  k_cvt<<<(nwi + 255) / 256, 256, 0, stream>>>(Wi_b, WiB16, nwi);
  k_cvt<<<(nwi + 255) / 256, 256, 0, stream>>>(Wh_f, WhF16, nwi);
  k_cvt<<<(nwi + 255) / 256, 256, 0, stream>>>(Wh_b, WhB16, nwi);
  const int nwo = K_TAGS * HID;
  k_cvt<<<(nwo + 255) / 256, 256, 0, stream>>>(W_out, Wo16, nwo);
  k_bias<<<G4H / 256, 256, 0, stream>>>(bi_f, bh_f, biasF);
  k_bias<<<G4H / 256, 256, 0, stream>>>(bi_b, bh_b, biasB);

  // time-parallel input projections (WMMA)
  k_gemm_z<<<dim3(16, 128), 128, 0, stream>>>(X16, WiF16, biasF, Zf);
  k_gemm_z<<<dim3(16, 128), 128, 0, stream>>>(X16, WiB16, biasB, Zb);

  // serial recurrences, one WGP per direction, Wh resident in VGPRs
  k_lstm<<<2, 1024, 0, stream>>>(WhF16, WhB16, Zf, Zb, h0, c0, H16);

  // output projection (WMMA)
  k_feats<<<128, 64, 0, stream>>>(H16, Wo16, b_out, feats);

  // Viterbi decode + backtrack
  k_viterbi<<<1, 32, 0, stream>>>(feats, trans, (float*)d_out);
}
